// ResidualAttentionBlock_14130442403962
// MI455X (gfx1250) — compile-verified, matrix operand order checked by reading, not run on hardware
//
#include <hip/hip_runtime.h>
#include <hip/hip_bf16.h>

typedef __attribute__((ext_vector_type(16))) _Float16 v16h;
typedef __attribute__((ext_vector_type(8)))  _Float16 h8;
typedef __attribute__((ext_vector_type(8)))  float    v8f;
typedef __attribute__((ext_vector_type(4)))  int      i4;

typedef __attribute__((address_space(1))) i4 g_i4;   // global int4
typedef __attribute__((address_space(3))) i4 l_i4;   // LDS int4

#define B_   4
#define N_   744
#define W_   128
#define H_   8
#define S_   768            // padded sequence (multiple of 32) for WMMA tiling
#define J_   3072           // 3*H*W
#define SCALE 0.08838834764831845f   // 1/sqrt(128)
#define SUNH  4.5f

// workspace byte offsets (all 16B aligned); total = 103,342,080 B (< 192MB L2)
#define OFF_XH   0u           // fp16 x copy              [B,N,W]        761856 B
#define OFF_WH   761856u      // fp16 wqv_w copy          [3072,128]     786432 B
#define OFF_Q    1548288u     // fp32 q                   [B,N,H,W]    12189696 B
#define OFF_VFT  13737984u    // fp16 vf transposed       [B,H,W,S]     6291456 B
#define OFF_VBT  20029440u    // fp16 vb transposed       [B,H,W,S]     6291456 B
#define OFF_P    26320896u    // fp16 probs  p[i=s][j=d]  [B,H,S,S]    37748736 B
#define OFF_PT   64069632u    // fp16 probs^T pT[d][s]    [B,H,S,S]    37748736 B
#define OFF_Y    101818368u   // fp32 y = sum_h(bf+bb)    [B,N,W]       1523712 B

#define HAS_ASYNC_LDS __has_builtin(__builtin_amdgcn_global_load_async_to_lds_b128)

__device__ __forceinline__ void wait_async_le1() {
#if __has_builtin(__builtin_amdgcn_s_wait_asynccnt)
    __builtin_amdgcn_s_wait_asynccnt(1);
#else
    asm volatile("s_wait_asynccnt 0x1" ::: "memory");
#endif
}
__device__ __forceinline__ void wait_async_le0() {
#if __has_builtin(__builtin_amdgcn_s_wait_asynccnt)
    __builtin_amdgcn_s_wait_asynccnt(0);
#else
    asm volatile("s_wait_asynccnt 0x0" ::: "memory");
#endif
}

// ---------------- init: fp16 conversions + zero padded buffers --------------
__global__ void k_init(const float* __restrict__ x, const float* __restrict__ wqv,
                       _Float16* __restrict__ xh, _Float16* __restrict__ wh,
                       _Float16* __restrict__ vfT, _Float16* __restrict__ vbT,
                       _Float16* __restrict__ p, _Float16* __restrict__ pT) {
    const long NXH = (long)B_ * N_ * W_;       // 380928
    const long NWH = (long)J_ * W_;            // 393216
    const long NV  = (long)B_ * H_ * S_ * W_;  // 3145728
    const long NP  = (long)B_ * H_ * S_ * S_;  // 18874368
    long i      = (long)blockIdx.x * blockDim.x + threadIdx.x;
    long stride = (long)gridDim.x * blockDim.x;
    for (long t = i; t < NP; t += stride) {
        if (t < NXH) xh[t] = (_Float16)x[t];
        if (t < NWH) wh[t] = (_Float16)wqv[t];
        if (t < NV) { vfT[t] = (_Float16)0.f; vbT[t] = (_Float16)0.f; }
        p[t]  = (_Float16)0.f;
        pT[t] = (_Float16)0.f;
    }
}

// ---------------- QV projection GEMM: v = x @ wqv^T + b (WMMA) --------------
// one wave per 16x16 output tile; M = B*N = 2976 (186 tiles), Ndim = 3072 (192)
__global__ void __launch_bounds__(32)
k_qv_gemm(const _Float16* __restrict__ xh, const _Float16* __restrict__ wh,
          const float* __restrict__ bias, float* __restrict__ q,
          _Float16* __restrict__ vfT, _Float16* __restrict__ vbT) {
    const int mt = blockIdx.x;          // 0..185
    const int nt = blockIdx.y;          // 0..191
    const int lane = threadIdx.x & 31;
    const int r = lane & 15, hb = lane >> 4;
    const int mrow = mt * 16 + r;       // A row held by this lane
    const int jcol = nt * 16 + r;       // B column held by this lane
    v8f c = {};
    for (int kt = 0; kt < W_ / 32; ++kt) {
        const int k0 = kt * 32 + hb * 8;
        h8 a0 = *(const h8*)(xh + (long)mrow * W_ + k0);
        h8 a1 = *(const h8*)(xh + (long)mrow * W_ + k0 + 16);
        h8 b0 = *(const h8*)(wh + (long)jcol * W_ + k0);      // B[k][j]=wqv[j][k]
        h8 b1 = *(const h8*)(wh + (long)jcol * W_ + k0 + 16);
        v16h a, bm;
        #pragma unroll
        for (int e = 0; e < 8; ++e) { a[e]=a0[e]; a[e+8]=a1[e]; bm[e]=b0[e]; bm[e+8]=b1[e]; }
        c = __builtin_amdgcn_wmma_f32_16x16x32_f16(false, a, false, bm,
                                                   (short)0, c, false, false);
    }
    // route: heads 0..7 -> q fp32 [b,s,h,w];  8..15 -> vfT fp16 [b,h,w,s];
    //        16..23 -> vbT fp16 [b,h,w,s]   (transposed so GEMM2 B-ops are K-contig)
    #pragma unroll
    for (int e = 0; e < 8; ++e) {
        const int m = mt * 16 + hb * 8 + e;     // < 2976 always
        const int j = nt * 16 + r;
        const float val = c[e] + bias[j];
        const int b = m / N_, s = m % N_;
        const int head = j >> 7, w = j & 127;
        if (head < H_) {
            q[(((long)(b * N_ + s)) * H_ + head) * W_ + w] = val;
        } else if (head < 2 * H_) {
            vfT[((long)(b * H_ + (head - H_)) * W_ + w) * S_ + s] = (_Float16)val;
        } else {
            vbT[((long)(b * H_ + (head - 2 * H_)) * W_ + w) * S_ + s] = (_Float16)val;
        }
    }
}

// ---------------- L1 scores + softmax over s (VALU bottleneck) --------------
// scores are always <= 0, so the appended zero logit is the max:
//   p[s,d] = exp(a[s,d]) / (1 + sum_s exp(a[s,d]))
// q rows are double-buffered into LDS via GLOBAL_LOAD_ASYNC_TO_LDS_B128
// (ASYNCcnt pipelining: issue row s+1, wait <=1 -> row s landed, compute).
#define DT 64
__global__ void __launch_bounds__(64)
k_scores(const float* __restrict__ x, const float* __restrict__ wk,
         const float* __restrict__ q, _Float16* __restrict__ p,
         _Float16* __restrict__ pT) {
    __shared__ float kt[DT][W_ + 1];    // stride 129 -> conflict-free (129 % 64 == 1)
    __shared__ float qs2[2][W_];
    const int tid = threadIdx.x;        // 0..63
    const int bh  = blockIdx.y;         // 0..31
    const int b = bh / H_, h = bh % H_;
    const int d0 = blockIdx.x * DT;     // 12 tiles cover S_=768
    const int d  = d0 + tid;
    const bool valid = (d < N_);

    for (int idx = tid; idx < DT * W_; idx += 64) {
        const int dl = idx >> 7, w = idx & 127;
        const int dg = d0 + dl;
        kt[dl][w] = (dg < N_) ? x[((long)b * N_ + dg) * W_ + w] * wk[h * W_ + w] : 0.f;
    }
    __syncthreads();

    const float* qb = q + (long)b * N_ * H_ * W_ + (long)h * W_;  // row s: qb + s*H*W

#if HAS_ASYNC_LDS
    if (tid < 32)   // one wave DMAs the 512B row: 32 lanes x b128
        __builtin_amdgcn_global_load_async_to_lds_b128(
            (g_i4*)(qb + tid * 4), (l_i4*)&qs2[0][tid * 4], 0, 0);
#endif

    float colsum = 0.f;
    for (int s = 0; s < N_; ++s) {
        const float* qs = qs2[s & 1];
#if HAS_ASYNC_LDS
        if (s + 1 < N_ && tid < 32)
            __builtin_amdgcn_global_load_async_to_lds_b128(
                (g_i4*)(qb + (long)(s + 1) * H_ * W_ + tid * 4),
                (l_i4*)&qs2[(s + 1) & 1][tid * 4], 0, 0);
        if (s + 1 < N_) wait_async_le1(); else wait_async_le0();
        __syncthreads();                 // row s visible to both waves
#else
        __syncthreads();
        #pragma unroll
        for (int i = 0; i < 2; ++i)
            qs2[s & 1][tid + 64 * i] = qb[(long)s * H_ * W_ + tid + 64 * i];
        __syncthreads();
#endif
        float acc = 0.f;
        #pragma unroll 8
        for (int w = 0; w < W_; ++w) acc += fabsf(qs[w] - kt[tid][w]);
        const float e = valid ? __expf(-acc * SCALE) : 0.f;
        colsum += e;
        p[((long)bh * S_ + s) * S_ + d] = (_Float16)e;
        if (valid) pT[((long)bh * S_ + d) * S_ + s] = (_Float16)e;
#if HAS_ASYNC_LDS
        __syncthreads();                 // all reads done before next DMA overwrites
#endif
    }
    if (valid) {
        const float rn = 1.f / (1.f + colsum);
        for (int s = 0; s < N_; ++s) {
            const long ip = ((long)bh * S_ + s) * S_ + d;
            const long it = ((long)bh * S_ + d) * S_ + s;
            p[ip]  = (_Float16)((float)p[ip] * rn);
            pT[it] = (_Float16)((float)pT[it] * rn);
        }
    }
}

// ---------------- attention apply: y[b,d,w] = sum_h (P^T Vf + P Vb) (WMMA) --
// one wave per 32x32 output block (2x2 register tiles); every fragment is
// K-contiguous -> exactly two b128 loads; 16 b128 loads : 8 WMMAs per K-step;
// fuses both einsums and the head reduction (1536 WMMAs/wave).
__global__ void __launch_bounds__(32)
k_apply(const _Float16* __restrict__ p, const _Float16* __restrict__ pT,
        const _Float16* __restrict__ vfT, const _Float16* __restrict__ vbT,
        float* __restrict__ y) {
    const int wt2 = blockIdx.x;         // 0..3  (pairs of w-tiles)
    const int dt2 = blockIdx.y;         // 0..23 (pairs of d-tiles)
    const int b   = blockIdx.z;         // 0..3
    const int lane = threadIdx.x & 31;
    const int r = lane & 15, hb = lane >> 4;
    int drow[2], wcol[2];
    drow[0] = dt2 * 32 + r;  drow[1] = drow[0] + 16;
    wcol[0] = wt2 * 32 + r;  wcol[1] = wcol[0] + 16;
    v8f c[2][2] = {};
    for (int h = 0; h < H_; ++h) {
        const _Float16* pb  = p   + ((long)(b * H_ + h)) * S_ * S_;
        const _Float16* pbT = pT  + ((long)(b * H_ + h)) * S_ * S_;
        const _Float16* vf  = vfT + ((long)(b * H_ + h)) * W_ * S_;
        const _Float16* vb  = vbT + ((long)(b * H_ + h)) * W_ * S_;
        for (int ktile = 0; ktile < S_ / 32; ++ktile) {
            const int s0 = ktile * 32 + hb * 8;
            __builtin_prefetch(pbT + (long)drow[0] * S_ + s0 + 64, 0, 1);
            __builtin_prefetch(pb  + (long)drow[0] * S_ + s0 + 64, 0, 1);
            v16h abf[2], abb[2], bvf[2], bvb[2];
            #pragma unroll
            for (int i = 0; i < 2; ++i) {
                h8 f0 = *(const h8*)(pbT + (long)drow[i] * S_ + s0);
                h8 f1 = *(const h8*)(pbT + (long)drow[i] * S_ + s0 + 16);
                h8 g0 = *(const h8*)(pb  + (long)drow[i] * S_ + s0);
                h8 g1 = *(const h8*)(pb  + (long)drow[i] * S_ + s0 + 16);
                h8 u0 = *(const h8*)(vf  + (long)wcol[i] * S_ + s0);
                h8 u1 = *(const h8*)(vf  + (long)wcol[i] * S_ + s0 + 16);
                h8 w0 = *(const h8*)(vb  + (long)wcol[i] * S_ + s0);
                h8 w1 = *(const h8*)(vb  + (long)wcol[i] * S_ + s0 + 16);
                #pragma unroll
                for (int e = 0; e < 8; ++e) {
                    abf[i][e] = f0[e]; abf[i][e + 8] = f1[e];
                    abb[i][e] = g0[e]; abb[i][e + 8] = g1[e];
                    bvf[i][e] = u0[e]; bvf[i][e + 8] = u1[e];
                    bvb[i][e] = w0[e]; bvb[i][e + 8] = w1[e];
                }
            }
            #pragma unroll
            for (int i = 0; i < 2; ++i) {
                #pragma unroll
                for (int j = 0; j < 2; ++j) {
                    c[i][j] = __builtin_amdgcn_wmma_f32_16x16x32_f16(
                        false, abf[i], false, bvf[j], (short)0, c[i][j], false, false);
                    c[i][j] = __builtin_amdgcn_wmma_f32_16x16x32_f16(
                        false, abb[i], false, bvb[j], (short)0, c[i][j], false, false);
                }
            }
        }
    }
    #pragma unroll
    for (int i = 0; i < 2; ++i) {
        #pragma unroll
        for (int j = 0; j < 2; ++j) {
            #pragma unroll
            for (int e = 0; e < 8; ++e) {
                const int dd = dt2 * 32 + i * 16 + hb * 8 + e;
                if (dd < N_)
                    y[((long)b * N_ + dd) * W_ + wt2 * 32 + j * 16 + r] = c[i][j][e];
            }
        }
    }
}

// ---------------- epilogue: qgelu + fanin GEMM + residual -------------------
__global__ void __launch_bounds__(128)
k_epilogue(const float* __restrict__ x, const float* __restrict__ y,
           const float* __restrict__ fw, const float* __restrict__ fb,
           float* __restrict__ out) {
    __shared__ float z[W_];
    const int row = blockIdx.x;         // 0..B*N-1
    const int j   = threadIdx.x;        // 0..127
    const float v = y[(long)row * W_ + j] + SUNH;
    z[j] = v / (1.f + __expf(-1.702f * v)) - SUNH;   // qgelu(v) - SUN/2
    __syncthreads();
    const float* frow = fw + (long)j * W_;
    float acc = fb[j];
    #pragma unroll 4
    for (int k = 0; k < W_; ++k) acc += z[k] * frow[k];
    out[(long)row * W_ + j] = x[(long)row * W_ + j] + acc;
}

extern "C" void kernel_launch(void* const* d_in, const int* in_sizes, int n_in,
                              void* d_out, int out_size, void* d_ws, size_t ws_size,
                              hipStream_t stream) {
    const float* x    = (const float*)d_in[0];
    const float* wk   = (const float*)d_in[1];
    const float* wqv  = (const float*)d_in[2];
    const float* wqvb = (const float*)d_in[3];
    const float* fw   = (const float*)d_in[4];
    const float* fb   = (const float*)d_in[5];
    float* out = (float*)d_out;

    char* ws = (char*)d_ws;
    _Float16* xh  = (_Float16*)(ws + OFF_XH);
    _Float16* wh  = (_Float16*)(ws + OFF_WH);
    float*    q   = (float*)   (ws + OFF_Q);
    _Float16* vfT = (_Float16*)(ws + OFF_VFT);
    _Float16* vbT = (_Float16*)(ws + OFF_VBT);
    _Float16* p   = (_Float16*)(ws + OFF_P);
    _Float16* pT  = (_Float16*)(ws + OFF_PT);
    float*    y   = (float*)   (ws + OFF_Y);

    k_init<<<2048, 256, 0, stream>>>(x, wqv, xh, wh, vfT, vbT, p, pT);

    dim3 g1(186, 192);                       // [2976/16, 3072/16]
    k_qv_gemm<<<g1, 32, 0, stream>>>(xh, wh, wqvb, q, vfT, vbT);

    dim3 g2(S_ / DT, B_ * H_);               // [12, 32]
    k_scores<<<g2, DT, 0, stream>>>(x, wk, q, p, pT);

    dim3 g3(W_ / 32, S_ / 32, B_);           // [4, 24, 4]
    k_apply<<<g3, 32, 0, stream>>>(p, pT, vfT, vbT, y);

    k_epilogue<<<B_ * N_, W_, 0, stream>>>(x, y, fw, fb, out);
}
